// KMeans_5592047419506
// MI455X (gfx1250) — matrix operand order repaired and checked by reading, not compiled, not run on hardware
//
#include <hip/hip_runtime.h>
#include <hip/hip_fp16.h>
#include <hip/hip_bf16.h>

typedef __attribute__((ext_vector_type(16))) _Float16 v16h;
typedef __attribute__((ext_vector_type(8)))  _Float16 v8h;
typedef __attribute__((ext_vector_type(4)))  _Float16 v4h;
typedef __attribute__((ext_vector_type(8)))  float    v8f;

#define VQ_D 1024   // codebook / feature dim
#define VQ_K 1024   // number of codewords

// ---------------------------------------------------------------------------
// Kernel 1: codebook -> f16 copy, per-row squared norms, zero histogram bins.
// Grid: K blocks x 256 threads (each thread handles one float4 of a row).
// ---------------------------------------------------------------------------
__global__ __launch_bounds__(256)
void vq_prep(const float* __restrict__ cb, _Float16* __restrict__ cbh,
             float* __restrict__ cbnorm, int* __restrict__ counts) {
    __shared__ float red[256];
    const int row = blockIdx.x;
    const int t   = threadIdx.x;

    if (row == 0) {
        for (int i = t; i < VQ_K; i += 256) counts[i] = 0;
    }

    const float4* src = (const float4*)(cb + (size_t)row * VQ_D);
    float4 v = src[t];                       // D/4 == 256 -> one per thread
    v4h h = { (_Float16)v.x, (_Float16)v.y, (_Float16)v.z, (_Float16)v.w };
    *(v4h*)(cbh + (size_t)row * VQ_D + t * 4) = h;

    red[t] = v.x * v.x + v.y * v.y + v.z * v.z + v.w * v.w;
    __syncthreads();
    for (int off = 128; off > 0; off >>= 1) {
        if (t < off) red[t] += red[t + off];
        __syncthreads();
    }
    if (t == 0) cbnorm[row] = red[0];
}

// ---------------------------------------------------------------------------
// Kernel 2: WMMA score GEMM + argmax.
// Block = 256 threads (8 waves). Each block owns a 16-row tile of x.
// Wave w scans codeword tiles [w*8, w*8+8), processed as 2 groups of 4
// concurrent tiles so one A-fragment load feeds 4 WMMAs.
// Score(n) = 2 * <x_m, e_n> - ||e_n||^2  (row-constant ||x||^2 dropped).
// ---------------------------------------------------------------------------
__global__ __launch_bounds__(256)
void vq_argmax(const float* __restrict__ x, const _Float16* __restrict__ cbh,
               const float* __restrict__ cbnorm, int* __restrict__ idxOut,
               int* __restrict__ counts) {
    __shared__ __attribute__((aligned(32))) _Float16 As[16 * VQ_D]; // 32 KB
    __shared__ float sBest[8 * 16];
    __shared__ int   sIdx [8 * 16];

    const int tid     = threadIdx.x;
    const int lane    = tid & 31;
    const int wave    = __builtin_amdgcn_readfirstlane(tid >> 5); // scalar
    const int rowBase = blockIdx.x * 16;

    // ---- stage 16 rows of x into LDS as f16 (each block reads its rows once)
    {
        const float4* src = (const float4*)(x + (size_t)rowBase * VQ_D);
        for (int i = tid; i < 16 * VQ_D / 4; i += 256) {
            float4 v = src[i];
            v4h h = { (_Float16)v.x, (_Float16)v.y, (_Float16)v.z, (_Float16)v.w };
            *(v4h*)(&As[i * 4]) = h;
        }
    }
    __syncthreads();

    const int  col    = lane & 15;        // column-in-tile (B/C) & row M for A frag
    const bool hi     = (lane >= 16);
    const int  kAoff  = hi ? 16 : 0;      // A layout: halves own K 0-15 / 16-31 (matches B)
    const int  kBoff  = hi ? 16 : 0;      // B layout: halves own K rows 0-15 / 16-31

    float best[8];
    int   bidx[8];
#pragma unroll
    for (int r = 0; r < 8; ++r) { best[r] = -3.0e38f; bidx[r] = 0; }

    const _Float16* aRowPtr = &As[(size_t)col * VQ_D + kAoff];

    for (int g = 0; g < 2; ++g) {                      // 2 groups of 4 tiles
        const int n0 = (wave * 8 + g * 4) * 16 + col;  // lane's codeword, tile 0
        // tiles 1..3 are n0 + 16, +32, +48 -> rows n0+16*t of the codebook
        const _Float16* bp = cbh + (size_t)n0 * VQ_D + kBoff;

        v8f c0 = {}, c1 = {}, c2 = {}, c3 = {};
#pragma unroll 2
        for (int db = 0; db < VQ_D; db += 32) {
            union { v16h v; v8h h[2]; } a;
            a.h[0] = *(const v8h*)(aRowPtr + db);        // K = db+kAoff .. +7
            a.h[1] = *(const v8h*)(aRowPtr + db + 8);    // K = db+kAoff+8 .. +15
            v16h b0 = *(const v16h*)(bp + db);
            v16h b1 = *(const v16h*)(bp + db + (size_t)16 * VQ_D);
            v16h b2 = *(const v16h*)(bp + db + (size_t)32 * VQ_D);
            v16h b3 = *(const v16h*)(bp + db + (size_t)48 * VQ_D);
            c0 = __builtin_amdgcn_wmma_f32_16x16x32_f16(
                     false, a.v, false, b0, (short)0, c0, false, false);
            c1 = __builtin_amdgcn_wmma_f32_16x16x32_f16(
                     false, a.v, false, b1, (short)0, c1, false, false);
            c2 = __builtin_amdgcn_wmma_f32_16x16x32_f16(
                     false, a.v, false, b2, (short)0, c2, false, false);
            c3 = __builtin_amdgcn_wmma_f32_16x16x32_f16(
                     false, a.v, false, b3, (short)0, c3, false, false);
        }

        // fold the 4 tiles into the running per-lane argmax (ascending n keeps
        // first-occurrence tie-breaking with strict '>')
        const float nn0 = cbnorm[n0];
        const float nn1 = cbnorm[n0 + 16];
        const float nn2 = cbnorm[n0 + 32];
        const float nn3 = cbnorm[n0 + 48];
#pragma unroll
        for (int r = 0; r < 8; ++r) {
            float s0 = 2.0f * c0[r] - nn0;
            float s1 = 2.0f * c1[r] - nn1;
            float s2 = 2.0f * c2[r] - nn2;
            float s3 = 2.0f * c3[r] - nn3;
            if (s0 > best[r]) { best[r] = s0; bidx[r] = n0;      }
            if (s1 > best[r]) { best[r] = s1; bidx[r] = n0 + 16; }
            if (s2 > best[r]) { best[r] = s2; bidx[r] = n0 + 32; }
            if (s3 > best[r]) { best[r] = s3; bidx[r] = n0 + 48; }
        }
    }

    // ---- cross-lane argmax within each 16-lane half (rows 0-7 / 8-15)
#pragma unroll
    for (int r = 0; r < 8; ++r) {
        float b0 = best[r];
        int   i0 = bidx[r];
        for (int off = 1; off < 16; off <<= 1) {
            float ob = __shfl_xor(b0, off, 32);
            int   oi = __shfl_xor(i0, off, 32);
            if (ob > b0 || (ob == b0 && oi < i0)) { b0 = ob; i0 = oi; }
        }
        best[r] = b0;
        bidx[r] = i0;
    }

    // ---- cross-wave reduction via LDS
    if (col == 0) {
        const int rbase = hi ? 8 : 0;
#pragma unroll
        for (int r = 0; r < 8; ++r) {
            sBest[wave * 16 + rbase + r] = best[r];
            sIdx [wave * 16 + rbase + r] = bidx[r];
        }
    }
    __syncthreads();

    if (tid < 16) {
        float b0 = sBest[tid];
        int   i0 = sIdx[tid];
        for (int w = 1; w < 8; ++w) {
            float ob = sBest[w * 16 + tid];
            int   oi = sIdx [w * 16 + tid];
            if (ob > b0 || (ob == b0 && oi < i0)) { b0 = ob; i0 = oi; }
        }
        idxOut[rowBase + tid] = i0;
        atomicAdd(&counts[i0], 1);
    }
}

// ---------------------------------------------------------------------------
// Kernel 3: gather codeword + squared error (pure bandwidth, float4).
// Grid: N blocks x 256 threads (one row per block, 4 floats per thread).
// ---------------------------------------------------------------------------
__global__ __launch_bounds__(256)
void vq_gather(const float* __restrict__ x, const float* __restrict__ cb,
               const int* __restrict__ idx, float* __restrict__ outQ,
               float* __restrict__ outL) {
    const int row  = blockIdx.x;
    const int t    = threadIdx.x;
    const int code = idx[row];

    const float4* cbp = (const float4*)(cb + (size_t)code * VQ_D);
    const float4* xp  = (const float4*)(x  + (size_t)row  * VQ_D);
    float4*       qp  = (float4*)(outQ + (size_t)row * VQ_D);
    float4*       lp  = (float4*)(outL + (size_t)row * VQ_D);

    float4 q = cbp[t];
    float4 v = xp[t];
    float4 l;
    l.x = (q.x - v.x) * (q.x - v.x);
    l.y = (q.y - v.y) * (q.y - v.y);
    l.z = (q.z - v.z) * (q.z - v.z);
    l.w = (q.w - v.w) * (q.w - v.w);
    qp[t] = q;
    lp[t] = l;
}

// ---------------------------------------------------------------------------
// Kernel 4: perplexity over codeword usage. One block.
// ---------------------------------------------------------------------------
__global__ __launch_bounds__(256)
void vq_perp(const int* __restrict__ counts, float* __restrict__ out, float invN) {
    __shared__ float red[256];
    const int t = threadIdx.x;
    float s = 0.0f;
    for (int i = t; i < VQ_K; i += 256) {
        float p = (float)counts[i] * invN;
        s += p * logf(p + 1e-10f);
    }
    red[t] = s;
    __syncthreads();
    for (int off = 128; off > 0; off >>= 1) {
        if (t < off) red[t] += red[t + off];
        __syncthreads();
    }
    if (t == 0) out[0] = expf(-red[0]);
}

// ---------------------------------------------------------------------------
extern "C" void kernel_launch(void* const* d_in, const int* in_sizes, int n_in,
                              void* d_out, int out_size, void* d_ws, size_t ws_size,
                              hipStream_t stream) {
    (void)n_in; (void)out_size; (void)ws_size;

    const float* x  = (const float*)d_in[0];   // (16, 4096, 1024) f32
    const float* cb = (const float*)d_in[1];   // (1024, 1024) f32
    const int N = in_sizes[0] / VQ_D;          // 65536 query vectors

    // workspace layout (d_ws is 256B aligned from hipMalloc)
    char* ws = (char*)d_ws;
    _Float16* cbh    = (_Float16*)ws;                                  // 2 MB
    float*    cbnorm = (float*)(ws + (size_t)VQ_K * VQ_D * sizeof(_Float16));
    int*      counts = (int*)((char*)cbnorm + VQ_K * sizeof(float));
    int*      idx    = (int*)((char*)counts + VQ_K * sizeof(int));     // N ints

    float* outQ = (float*)d_out;                    // quantize
    float* outL = outQ + (size_t)N * VQ_D;          // codebook_loss
    float* outP = outL + (size_t)N * VQ_D;          // perplexity scalar

    vq_prep  <<<VQ_K,   256, 0, stream>>>(cb, cbh, cbnorm, counts);
    vq_argmax<<<N / 16, 256, 0, stream>>>(x, cbh, cbnorm, idx, counts);
    vq_gather<<<N,      256, 0, stream>>>(x, cb, idx, outQ, outL);
    vq_perp  <<<1,      256, 0, stream>>>(counts, outP, 1.0f / (float)N);
}
